// GCNModel_55817394978866
// MI455X (gfx1250) — matrix operand order repaired and checked by reading, not compiled, not run on hardware
//
#include <hip/hip_runtime.h>
#include <hip/hip_bf16.h>
#include <math.h>

typedef __attribute__((ext_vector_type(16))) _Float16 v16h;
typedef __attribute__((ext_vector_type(8)))  float    v8f;

// ---------------------------------------------------------------------------
// Zero a float region (atomic accumulation targets must start at 0 each call).
// ---------------------------------------------------------------------------
__global__ void zero_f32_kernel(float* __restrict__ p, long n) {
    long i = (long)blockIdx.x * blockDim.x + threadIdx.x;
    long stride = (long)gridDim.x * blockDim.x;
    for (; i < n; i += stride) p[i] = 0.0f;
}

// ---------------------------------------------------------------------------
// Degree: deg[dst[e]] += 1
// ---------------------------------------------------------------------------
__global__ void degree_kernel(const int* __restrict__ dst, float* __restrict__ deg, int E) {
    int e = blockIdx.x * blockDim.x + threadIdx.x;
    if (e < E) atomicAdd(&deg[dst[e]], 1.0f);
}

// ---------------------------------------------------------------------------
// Edge norm: rsqrt(max(deg[src],1)) * rsqrt(max(deg[dst],1))
// ---------------------------------------------------------------------------
__global__ void norm_kernel(const int* __restrict__ src, const int* __restrict__ dst,
                            const float* __restrict__ deg, float* __restrict__ norm, int E) {
    int e = blockIdx.x * blockDim.x + threadIdx.x;
    if (e < E) {
        float ds = fmaxf(deg[src[e]], 1.0f);
        float dd = fmaxf(deg[dst[e]], 1.0f);
        norm[e] = rsqrtf(ds) * rsqrtf(dd);
    }
}

// ---------------------------------------------------------------------------
// Aggregation: one wave per edge; lanes carry float4 chunks of the feature
// row -> coalesced gather, per-float f32 global atomics for the scatter
// (random traffic resolves in the 192 MB L2; x/h1 fit entirely).
// ---------------------------------------------------------------------------
__global__ void aggregate_kernel(const float* __restrict__ X,
                                 const int*   __restrict__ src,
                                 const int*   __restrict__ dst,
                                 const float* __restrict__ norm,
                                 float*       __restrict__ agg,
                                 int E, int F) {
    const int wavesPerBlock = blockDim.x >> 5;
    const int lane = threadIdx.x & 31;
    int e = blockIdx.x * wavesPerBlock + (threadIdx.x >> 5);
    if (e >= E) return;
    const int s = src[e];
    const int d = dst[e];
    const float nv = norm[e];
    const float4* xr = (const float4*)(X + (long)s * F);
    float* outr = agg + (long)d * F;
    const int fper = F >> 2;
    for (int c = lane; c < fper; c += 32) {
        float4 xv = xr[c];
        float* out = outr + (long)c * 4;
        atomicAdd(out + 0, xv.x * nv);
        atomicAdd(out + 1, xv.y * nv);
        atomicAdd(out + 2, xv.z * nv);
        atomicAdd(out + 3, xv.w * nv);
    }
}

// ---------------------------------------------------------------------------
// Pack A (f32 [M,K] row-major) -> f16 fragments in WMMA A-layout.
// Packed block (tm, kb) = 512 f16; lane L owns elements [L*16 .. L*16+15].
// Element j maps to K = kb*32 + (j>>3)*16 + half*8 + ((j>>1)&3)*2 + (j&1),
// row = tm*16 + L%16.
// ---------------------------------------------------------------------------
__global__ void pack_a_kernel(const float* __restrict__ A, _Float16* __restrict__ Ap,
                              int M, int K) {
    const int kBlocks = K >> 5;
    long total = (long)M * K;
    long i = (long)blockIdx.x * blockDim.x + threadIdx.x;
    long stride = (long)gridDim.x * blockDim.x;
    for (; i < total; i += stride) {
        long blk    = i >> 9;           // /512
        int  within = (int)(i & 511);
        int  lane = within >> 4;
        int  j    = within & 15;
        int  tm = (int)(blk / kBlocks);
        int  kb = (int)(blk - (long)tm * kBlocks);
        int  half = lane >> 4;
        int  row  = tm * 16 + (lane & 15);
        int  k = kb * 32 + ((j >> 3) << 4) + (half << 3) + (((j >> 1) & 3) << 1) + (j & 1);
        Ap[i] = (_Float16)A[(long)row * K + k];
    }
}

// ---------------------------------------------------------------------------
// Pack W (f32 [K,N] row-major) -> f16 fragments in WMMA B-layout, with the
// tile count padded to tilesNpad (multiple of 4) so the GEMM runs fixed 1x4
// strips with no tail logic.  Lane L element j maps to k = kb*32 + (j>>3)*16 + half*8 + (j&7),
// col = tn*16 + L%16 (zero-filled beyond N).
// ---------------------------------------------------------------------------
__global__ void pack_b_kernel(const float* __restrict__ W, _Float16* __restrict__ Bp,
                              int K, int N, int tilesNpad) {
    const int kBlocks = K >> 5;
    long total = (long)tilesNpad * kBlocks * 512;
    long i = (long)blockIdx.x * blockDim.x + threadIdx.x;
    long stride = (long)gridDim.x * blockDim.x;
    for (; i < total; i += stride) {
        long blk    = i >> 9;
        int  within = (int)(i & 511);
        int  lane = within >> 4;
        int  j    = within & 15;
        int  tn = (int)(blk / kBlocks);
        int  kb = (int)(blk - (long)tn * kBlocks);
        int  half = lane >> 4;
        int  col  = tn * 16 + (lane & 15);
        int  k = kb * 32 + ((j >> 3) << 4) + (half << 3) + (j & 7);
        Bp[i] = (col < N) ? (_Float16)W[(long)k * N + col] : (_Float16)0.0f;
    }
}

// ---------------------------------------------------------------------------
// WMMA GEMM on pre-packed fragments: Out[M,N] = act(A @ W + bias).
// One wave computes a 16x64 strip (1 A tile row x 4 N-tiles): per k-step,
// 1 A fragment (2x b128) + 4 B fragments (8x b128) feed 4 back-to-back
// v_wmma_f32_16x16x32_f16 -> 4x A-fragment reuse, no divergence at the WMMAs.
// ---------------------------------------------------------------------------
__global__ void gemm_packed_wmma(const _Float16* __restrict__ Ap,
                                 const _Float16* __restrict__ Bp,
                                 const float*    __restrict__ bias,
                                 float*          __restrict__ Out,
                                 int M, int K, int N, int tilesNpad, int doRelu) {
    const int lane = threadIdx.x & 31;
    const int wavesPerBlock = blockDim.x >> 5;
    const int tilesM  = M >> 4;            // M multiple of 16
    const int kBlocks = K >> 5;            // K multiple of 32
    const int groupsN = tilesNpad >> 2;    // 4 N-tiles per wave

    long stripId = (long)blockIdx.x * wavesPerBlock + (threadIdx.x >> 5);
    if (stripId >= (long)tilesM * groupsN) return;   // wave-uniform exit

    const int tm = (int)(stripId / groupsN);
    const int tg = (int)(stripId - (long)tm * groupsN);
    const int tn0 = tg << 2;

    const long fragStride = (long)kBlocks * 32;      // v16h units per tile
    const v16h* Af  = (const v16h*)Ap + (long)tm * fragStride + lane;
    const v16h* Bf0 = (const v16h*)Bp + (long)(tn0 + 0) * fragStride + lane;
    const v16h* Bf1 = (const v16h*)Bp + (long)(tn0 + 1) * fragStride + lane;
    const v16h* Bf2 = (const v16h*)Bp + (long)(tn0 + 2) * fragStride + lane;
    const v16h* Bf3 = (const v16h*)Bp + (long)(tn0 + 3) * fragStride + lane;

    v8f c0 = {}, c1 = {}, c2 = {}, c3 = {};
#pragma unroll 2
    for (int kb = 0; kb < kBlocks; ++kb) {
        const long o = (long)kb * 32;
        v16h a  = Af[o];
        v16h b0 = Bf0[o];
        v16h b1 = Bf1[o];
        v16h b2 = Bf2[o];
        v16h b3 = Bf3[o];
        c0 = __builtin_amdgcn_wmma_f32_16x16x32_f16(false, a, false, b0, (short)0, c0, false, false);
        c1 = __builtin_amdgcn_wmma_f32_16x16x32_f16(false, a, false, b1, (short)0, c1, false, false);
        c2 = __builtin_amdgcn_wmma_f32_16x16x32_f16(false, a, false, b2, (short)0, c2, false, false);
        c3 = __builtin_amdgcn_wmma_f32_16x16x32_f16(false, a, false, b3, (short)0, c3, false, false);
    }

    // C/D layout: VGPR v holds (M = tm*16 + half*8 + v, N = tn*16 + lane%16)
    const int half = lane >> 4;
    const int baseRow = tm * 16 + (half << 3);
    v8f acc[4] = {c0, c1, c2, c3};
#pragma unroll
    for (int t = 0; t < 4; ++t) {
        const int colB = (tn0 + t) * 16 + (lane & 15);
        if (colB < N) {
            const float bv = bias[colB];
#pragma unroll
            for (int v = 0; v < 8; ++v) {
                float val = acc[t][v] + bv;
                if (doRelu) val = fmaxf(val, 0.0f);
                Out[(long)(baseRow + v) * N + colB] = val;
            }
        }
    }
}

// ---------------------------------------------------------------------------
// Row softmax over C columns; one wave32 per row, shfl_xor reductions.
// ---------------------------------------------------------------------------
__global__ void softmax_rows_kernel(const float* __restrict__ logits,
                                    float* __restrict__ out, int M, int C) {
    int row = blockIdx.x * (blockDim.x >> 5) + (threadIdx.x >> 5);
    if (row >= M) return;
    int lane = threadIdx.x & 31;
    const float* lr = logits + (long)row * C;

    float m = -INFINITY;
    for (int c = lane; c < C; c += 32) m = fmaxf(m, lr[c]);
#pragma unroll
    for (int off = 16; off > 0; off >>= 1) m = fmaxf(m, __shfl_xor(m, off, 32));

    float s = 0.0f;
    for (int c = lane; c < C; c += 32) s += __expf(lr[c] - m);
#pragma unroll
    for (int off = 16; off > 0; off >>= 1) s += __shfl_xor(s, off, 32);

    float inv = 1.0f / s;
    for (int c = lane; c < C; c += 32)
        out[(long)row * C + c] = __expf(lr[c] - m) * inv;
}

// ---------------------------------------------------------------------------
// Host-side launcher
// ---------------------------------------------------------------------------
static inline int grid_for(long n, int t, int cap) {
    long g = (n + t - 1) / t;
    if (g > cap) g = cap;
    if (g < 1) g = 1;
    return (int)g;
}

extern "C" void kernel_launch(void* const* d_in, const int* in_sizes, int n_in,
                              void* d_out, int out_size, void* d_ws, size_t ws_size,
                              hipStream_t stream) {
    const float* x  = (const float*)d_in[0];
    const int*   ei = (const int*)d_in[1];
    const float* W1 = (const float*)d_in[2];
    const float* b1 = (const float*)d_in[3];
    const float* W2 = (const float*)d_in[4];
    const float* b2 = (const float*)d_in[5];
    const float* Wd = (const float*)d_in[6];
    const float* bd = (const float*)d_in[7];

    const int F = 128;
    const int N = in_sizes[0] / F;         // 50000
    const int E = in_sizes[1] / 2;         // 600000
    const int H = in_sizes[3];             // 256
    const int C = in_sizes[7];             // 40

    const int* src = ei;
    const int* dst = ei + E;

    // Float workspace layout
    float* ws    = (float*)d_ws;
    float* deg   = ws;                             // N
    float* norm  = deg  + N;                       // E
    float* agg1  = norm + E;                       // N*F
    float* h1    = agg1 + (long)N * F;             // N*H
    float* agg2  = h1   + (long)N * H;             // N*H
    float* h2    = agg2 + (long)N * H;             // N*H
    float* logit = h2   + (long)N * H;             // N*C
    // f16 packed fragment buffers (reused across the three GEMMs)
    _Float16* Apack = (_Float16*)(logit + (long)N * C);          // N*H halves max
    _Float16* Bpack = Apack + (long)N * H;                       // H*Hpad halves max

    const int T = 256;
    const int waves = T >> 5;

    // Zero atomic targets: contiguous [deg,norm,agg1) span + agg2
    zero_f32_kernel<<<grid_for((long)N + E + (long)N * F, T, 65535), T, 0, stream>>>(
        deg, (long)N + E + (long)N * F);
    zero_f32_kernel<<<grid_for((long)N * H, T, 65535), T, 0, stream>>>(agg2, (long)N * H);

    // Degree + norm
    degree_kernel<<<(E + T - 1) / T, T, 0, stream>>>(dst, deg, E);
    norm_kernel  <<<(E + T - 1) / T, T, 0, stream>>>(src, dst, deg, norm, E);

    // ---------------- Layer 1: agg1 = scatter(x*norm); h1 = relu(agg1@W1+b1)
    aggregate_kernel<<<(E + waves - 1) / waves, T, 0, stream>>>(x, src, dst, norm, agg1, E, F);
    {
        int tilesNpad = (((H + 15) >> 4) + 3) & ~3;
        pack_a_kernel<<<grid_for((long)N * F, T, 262144), T, 0, stream>>>(agg1, Apack, N, F);
        pack_b_kernel<<<grid_for((long)tilesNpad * (F >> 5) * 512, T, 65535), T, 0, stream>>>(
            W1, Bpack, F, H, tilesNpad);
        long strips = (long)(N >> 4) * (tilesNpad >> 2);
        gemm_packed_wmma<<<(int)((strips + waves - 1) / waves), T, 0, stream>>>(
            Apack, Bpack, b1, h1, N, F, H, tilesNpad, 1);
    }

    // ---------------- Layer 2: agg2 = scatter(h1*norm); h2 = relu(agg2@W2+b2)
    aggregate_kernel<<<(E + waves - 1) / waves, T, 0, stream>>>(h1, src, dst, norm, agg2, E, H);
    {
        int tilesNpad = (((H + 15) >> 4) + 3) & ~3;
        pack_a_kernel<<<grid_for((long)N * H, T, 262144), T, 0, stream>>>(agg2, Apack, N, H);
        pack_b_kernel<<<grid_for((long)tilesNpad * (H >> 5) * 512, T, 65535), T, 0, stream>>>(
            W2, Bpack, H, H, tilesNpad);
        long strips = (long)(N >> 4) * (tilesNpad >> 2);
        gemm_packed_wmma<<<(int)((strips + waves - 1) / waves), T, 0, stream>>>(
            Apack, Bpack, b2, h2, N, H, H, tilesNpad, 1);
    }

    // ---------------- Head: logit = h2@Wd+bd; out = softmax(logit)
    {
        int tilesNpad = (((C + 15) >> 4) + 3) & ~3;    // 3 -> 4 tiles (zero-padded)
        pack_a_kernel<<<grid_for((long)N * H, T, 262144), T, 0, stream>>>(h2, Apack, N, H);
        pack_b_kernel<<<grid_for((long)tilesNpad * (H >> 5) * 512, T, 65535), T, 0, stream>>>(
            Wd, Bpack, H, C, tilesNpad);
        long strips = (long)(N >> 4) * (tilesNpad >> 2);
        gemm_packed_wmma<<<(int)((strips + waves - 1) / waves), T, 0, stream>>>(
            Apack, Bpack, bd, logit, N, H, C, tilesNpad, 0);

        softmax_rows_kernel<<<(N + waves - 1) / waves, T, 0, stream>>>(
            logit, (float*)d_out, N, C);
    }
}
